// Mamba2_26371099197593
// MI455X (gfx1250) — matrix operand order repaired and checked by reading, not compiled, not run on hardware
//
#include <hip/hip_runtime.h>

// ---------------- problem constants ----------------
#define B_    4
#define L_    4096
#define DM    512            // d_model
#define DI    2048           // d_inner
#define NH    8              // heads
#define HD    256            // headdim
#define DS_   64             // d_state
#define CH_   256            // chunk length
#define NCH   16             // chunks per sequence
#define CONVD 2176           // conv dim = DI + 2*DS
#define TOKS  (B_ * L_)

// ---------------- WMMA types ----------------
typedef __attribute__((ext_vector_type(16))) __bf16        bf16x16;
typedef __attribute__((ext_vector_type(8)))  float         v8f;
typedef __attribute__((ext_vector_type(4)))  unsigned int  u32x4;

union Frag {
    bf16x16        v;
    unsigned short s[16];
    u32x4          q[2];
};

__device__ __forceinline__ v8f wmma_bf16(const Frag& a, const Frag& b, v8f c) {
    return __builtin_amdgcn_wmma_f32_16x16x32_bf16(false, a.v, false, b.v,
                                                   (short)0, c, false, false);
}

__device__ __forceinline__ unsigned short f32_to_bf16(float f) {
    union { float f; unsigned int u; } x; x.f = f;
    unsigned int u = x.u;
    u += 0x7FFFu + ((u >> 16) & 1u);      // round to nearest even
    return (unsigned short)(u >> 16);
}
__device__ __forceinline__ float bf16_to_f32(unsigned short h) {
    union { unsigned int u; float f; } x; x.u = ((unsigned int)h) << 16;
    return x.f;
}

// ---------------- conversion kernels ----------------
__global__ void cvt_bf16_kernel(const float* __restrict__ src,
                                unsigned short* __restrict__ dst, size_t n) {
    size_t i = (size_t)blockIdx.x * 256 + threadIdx.x;
    if (i < n) dst[i] = f32_to_bf16(src[i]);
}

// W is (K x N) row-major f32 -> Wt is (N x K) row-major bf16 (K-contiguous B-operand)
__global__ void transpose_bf16_kernel(const float* __restrict__ src,
                                      unsigned short* __restrict__ dst, int K, int N) {
    size_t i = (size_t)blockIdx.x * 256 + threadIdx.x;
    if (i < (size_t)K * N) {
        int k = (int)(i / N), n = (int)(i % N);
        dst[(size_t)n * K + k] = f32_to_bf16(src[i]);
    }
}

// ---------------- dt = softplus(inputs @ Wdt + bias) ----------------
__global__ __launch_bounds__(256) void dt_kernel(const float* __restrict__ in,
                                                 const float* __restrict__ Wdt,
                                                 const float* __restrict__ dtb,
                                                 float* __restrict__ dtw) {
    const int tok  = blockIdx.x;
    const int hh   = threadIdx.x >> 5;       // wave = head
    const int lane = threadIdx.x & 31;
    const float* row = in + (size_t)tok * DM;
    float s = 0.f;
    for (int k = lane; k < DM; k += 32) s += row[k] * Wdt[(size_t)k * NH + hh];
    for (int off = 16; off; off >>= 1) s += __shfl_xor(s, off, 32);
    if (lane == 0) {
        float v = s + dtb[hh];
        dtw[(size_t)tok * NH + hh] = (v > 20.f) ? v : log1pf(expf(v));
    }
}

// ---------------- bf16 WMMA GEMM: A(MxK) row-major, Bt(NxK) row-major ----------------
template <bool OUT_F32>
__global__ __launch_bounds__(256) void gemm_bf16(const unsigned short* __restrict__ A,
                                                 const unsigned short* __restrict__ Bt,
                                                 void* __restrict__ Out,
                                                 int M, int N, int K) {
    const int gw   = (int)((blockIdx.x * (size_t)blockDim.x + threadIdx.x) >> 5);
    const int lane = threadIdx.x & 31;
    const int nw   = N / 64;
    const int m0   = (gw / nw) * 16;
    const int n0   = (gw % nw) * 64;
    if (m0 >= M) return;
    const int hl = lane >> 4, lm = lane & 15;

    v8f acc[4] = {};
    const unsigned short* Arow = A + (size_t)(m0 + lm) * K + hl * 8;
#pragma unroll 2
    for (int k0 = 0; k0 < K; k0 += 32) {
        if (k0 + 32 < K) {  // keep the HBM stream ahead of the WMMA chain
            __builtin_prefetch(Arow + k0 + 32, 0, 1);
            __builtin_prefetch(Bt + (size_t)(n0 + lm) * K + k0 + 32 + hl * 16, 0, 1);
        }
        Frag a;
        a.q[0] = *(const u32x4*)(Arow + k0);
        a.q[1] = *(const u32x4*)(Arow + k0 + 16);
#pragma unroll
        for (int j = 0; j < 4; ++j) {
            const unsigned short* Brow = Bt + (size_t)(n0 + j * 16 + lm) * K + k0 + hl * 8;
            Frag bb;
            bb.q[0] = *(const u32x4*)(Brow);
            bb.q[1] = *(const u32x4*)(Brow + 16);
            acc[j] = wmma_bf16(a, bb, acc[j]);
        }
    }
#pragma unroll
    for (int j = 0; j < 4; ++j)
#pragma unroll
        for (int r = 0; r < 8; ++r) {
            const int m = m0 + r + hl * 8;
            const int n = n0 + j * 16 + lm;
            if (OUT_F32) ((float*)Out)[(size_t)m * N + n] = acc[j][r];
            else ((unsigned short*)Out)[(size_t)m * N + n] = f32_to_bf16(acc[j][r]);
        }
}

// ---------------- depthwise conv(4) + SiLU + routing ----------------
__global__ __launch_bounds__(256) void conv_silu_route_kernel(
    const unsigned short* __restrict__ xbc, const float* __restrict__ ck,
    const float* __restrict__ dtw,
    unsigned short* __restrict__ xdt,  // (B,H,P,L) bf16 : x * dt, transposed
    unsigned short* __restrict__ bg,   // (B*L, 64) row-major
    unsigned short* __restrict__ bt,   // (B,64,L) transposed
    unsigned short* __restrict__ cg)   // (B*L, 64) row-major
{
    size_t idx = (size_t)blockIdx.x * 256 + threadIdx.x;
    if (idx >= (size_t)TOKS * CONVD) return;
    const int    ch  = (int)(idx % CONVD);
    const size_t tok = idx / CONVD;
    const int    l   = (int)(tok % L_);
    const int    b   = (int)(tok / L_);

    float acc = 0.f;
#pragma unroll
    for (int w = 0; w < 4; ++w) {
        int ls = l - 3 + w;
        if (ls >= 0)
            acc += bf16_to_f32(xbc[(tok - (size_t)(3 - w)) * CONVD + ch]) * ck[w * CONVD + ch];
    }
    const float v = acc / (1.f + expf(-acc));   // SiLU

    if (ch < DI) {
        const int h = ch >> 8, p = ch & 255;
        const float dt = dtw[tok * NH + h];
        xdt[(((size_t)(b * NH + h)) * HD + p) * L_ + l] = f32_to_bf16(v * dt);
    } else if (ch < DI + DS_) {
        const int n = ch - DI;
        const unsigned short hv = f32_to_bf16(v);
        bg[tok * DS_ + n] = hv;
        bt[((size_t)(b * DS_ + n)) * L_ + l] = hv;
    } else {
        const int n = ch - DI - DS_;
        cg[tok * DS_ + n] = f32_to_bf16(v);
    }
}

// ---------------- SSD chunk kernel: one block per (b, chunk, head) ----------------
__global__ __launch_bounds__(256) void ssd_chunk_kernel(
    const float* __restrict__ dtw, const float* __restrict__ alog,
    const unsigned short* __restrict__ bt, const unsigned short* __restrict__ bg,
    const unsigned short* __restrict__ cg, const unsigned short* __restrict__ xdt,
    unsigned short* __restrict__ ybf, float* __restrict__ Sloc,
    float* __restrict__ cumA_ws, float* __restrict__ Asum)
{
    __shared__ float sCum[CH_];
    __shared__ float sDec[CH_];
    // 32 KB buffer: first used to stage decayed b^T (64 x 256), then masked-G strips
    __shared__ __attribute__((aligned(16))) unsigned short sG[64 * CH_];

    const int tid  = threadIdx.x;
    const int lane = tid & 31;
    const int wv   = tid >> 5;
    const int hl   = lane >> 4;
    const int lm   = lane & 15;

    const int h = blockIdx.x & 7;
    const int c = (blockIdx.x >> 3) & 15;
    const int b = blockIdx.x >> 7;
    const size_t tok0 = (size_t)b * L_ + (size_t)c * CH_;

    // ---- inclusive scan of a*dt over the chunk ----
    const float a_h = -expf(alog[h]);
    sCum[tid] = a_h * dtw[(tok0 + tid) * NH + h];
    __syncthreads();
    for (int off = 1; off < CH_; off <<= 1) {
        float v = (tid >= off) ? sCum[tid - off] : 0.f;
        __syncthreads();
        sCum[tid] += v;
        __syncthreads();
    }
    const float cumLast = sCum[CH_ - 1];
    sDec[tid] = expf(cumLast - sCum[tid]);                 // decay_states
    cumA_ws[(size_t)blockIdx.x * CH_ + tid] = sCum[tid];
    if (tid == 0) Asum[blockIdx.x] = cumLast;
    __syncthreads();

    // ---- stage decayed b^T into LDS: sG[n][s] = b[s][n] * dec[s] ----
    // thread tid owns column s = tid -> decay is a single register, loads/stores coalesced
    {
        const float mydec = sDec[tid];
        const unsigned short* btb = bt + ((size_t)(b * DS_)) * L_ + (size_t)c * CH_ + tid;
#pragma unroll
        for (int n = 0; n < DS_; ++n)
            sG[n * CH_ + tid] = f32_to_bf16(bf16_to_f32(btb[(size_t)n * L_]) * mydec);
    }
    __syncthreads();

    const unsigned short* Xbase = xdt + (((size_t)(b * NH + h)) * HD) * L_ + (size_t)c * CH_;

    // ---- local chunk states: S[n][p] = sum_s bd^T[n][s] * xd[s][p] (pure WMMA loop) ----
    {
        const int n0  = (wv & 3) * 16;
        const int pt0 = (wv >> 2) * 8;
        const unsigned short* Arow = sG + (n0 + lm) * CH_ + hl * 8;
        v8f acc[8] = {};
#pragma unroll 2
        for (int k0 = 0; k0 < CH_; k0 += 32) {
            Frag a;
            a.q[0] = *(const u32x4*)(Arow + k0);
            a.q[1] = *(const u32x4*)(Arow + k0 + 16);
#pragma unroll
            for (int j = 0; j < 8; ++j) {
                const int p = (pt0 + j) * 16 + lm;
                const unsigned short* Xr = Xbase + (size_t)p * L_ + k0 + hl * 8;
                Frag x;
                x.q[0] = *(const u32x4*)(Xr);
                x.q[1] = *(const u32x4*)(Xr + 16);
                acc[j] = wmma_bf16(a, x, acc[j]);
            }
        }
        float* Sl = Sloc + (size_t)blockIdx.x * (HD * DS_);
#pragma unroll
        for (int j = 0; j < 8; ++j) {
            const int p = (pt0 + j) * 16 + lm;
#pragma unroll
            for (int r = 0; r < 8; ++r)
                Sl[(size_t)p * DS_ + n0 + r + hl * 8] = acc[j][r];   // (p,n) layout
        }
    }
    __syncthreads();   // done reading staged bd^T; sG is reused for G strips below

    // ---- y_diag in 64-row strips: G = mask(C * B^T), Y = G * Xd ----
    const int lt  = wv & 3;
    const int st0 = (wv >> 2) * 8;
    for (int strip = 0; strip < 4; ++strip) {
        const int lbase = strip * 64 + lt * 16;
        {
            v8f acc[8] = {};
            const unsigned short* Crow = cg + (tok0 + lbase + lm) * DS_ + hl * 8;
#pragma unroll
            for (int k0 = 0; k0 < DS_; k0 += 32) {
                Frag a;
                a.q[0] = *(const u32x4*)(Crow + k0);
                a.q[1] = *(const u32x4*)(Crow + k0 + 16);
#pragma unroll
                for (int j = 0; j < 8; ++j) {
                    const int s = (st0 + j) * 16 + lm;
                    const unsigned short* Brow = bg + (tok0 + s) * DS_ + k0 + hl * 8;
                    Frag bb;
                    bb.q[0] = *(const u32x4*)(Brow);
                    bb.q[1] = *(const u32x4*)(Brow + 16);
                    acc[j] = wmma_bf16(a, bb, acc[j]);
                }
            }
#pragma unroll
            for (int j = 0; j < 8; ++j) {
                const int s = (st0 + j) * 16 + lm;
#pragma unroll
                for (int r = 0; r < 8; ++r) {
                    const int ll = lbase + r + hl * 8;
                    const float g = (ll >= s) ? acc[j][r] * expf(sCum[ll] - sCum[s]) : 0.f;
                    sG[(ll - strip * 64) * CH_ + s] = f32_to_bf16(g);
                }
            }
        }
        __syncthreads();
        {
            v8f acc[8] = {};
            const unsigned short* Grow = sG + (lt * 16 + lm) * CH_ + hl * 8;
#pragma unroll 2
            for (int k0 = 0; k0 < CH_; k0 += 32) {
                Frag a;
                a.q[0] = *(const u32x4*)(Grow + k0);
                a.q[1] = *(const u32x4*)(Grow + k0 + 16);
#pragma unroll
                for (int j = 0; j < 8; ++j) {
                    const int p = (st0 + j) * 16 + lm;
                    const unsigned short* Xr = Xbase + (size_t)p * L_ + k0 + hl * 8;
                    Frag x;
                    x.q[0] = *(const u32x4*)(Xr);
                    x.q[1] = *(const u32x4*)(Xr + 16);
                    acc[j] = wmma_bf16(a, x, acc[j]);
                }
            }
#pragma unroll
            for (int j = 0; j < 8; ++j) {
                const int p = (st0 + j) * 16 + lm;
#pragma unroll
                for (int r = 0; r < 8; ++r) {
                    const int ll = lbase + r + hl * 8;
                    ybf[(tok0 + ll) * DI + (size_t)h * HD + p] = f32_to_bf16(acc[j][r]);
                }
            }
        }
        __syncthreads();
    }
}

// ---------------- inter-chunk sequential state scan (f32 accum, bf16 out) ----------------
__global__ void state_scan_kernel(const float* __restrict__ Sloc,
                                  unsigned short* __restrict__ Sscan,
                                  const float* __restrict__ Asum) {
    const size_t PN = (size_t)HD * DS_;
    const size_t e  = (size_t)blockIdx.x * 256 + threadIdx.x;
    if (e >= (size_t)B_ * NH * PN) return;
    const size_t pn = e % PN;
    const int h = (int)((e / PN) % NH);
    const int b = (int)(e / (PN * NH));
    float S = 0.f;
    for (int c = 0; c < NCH; ++c) {
        const size_t idx = ((size_t)((b * NCH + c) * NH + h)) * PN + pn;
        Sscan[idx] = f32_to_bf16(S);                       // state entering chunk c
        S = expf(Asum[(b * NCH + c) * NH + h]) * S + Sloc[idx];
    }
}

// ---------------- y_off = C*S_prev * exp(cumA), add y_diag, SiLU(z) gate ----------------
__global__ __launch_bounds__(256) void yoff_gate_kernel(
    const unsigned short* __restrict__ cg, const unsigned short* __restrict__ Sscan,
    const float* __restrict__ cumA_ws, const unsigned short* __restrict__ zbf,
    unsigned short* __restrict__ ybf)
{
    __shared__ float sExp[CH_];
    const int tid  = threadIdx.x;
    const int lane = tid & 31;
    const int wv   = tid >> 5;
    const int hl   = lane >> 4;
    const int lm   = lane & 15;
    const int h = blockIdx.x & 7;
    const int c = (blockIdx.x >> 3) & 15;
    const int b = blockIdx.x >> 7;
    const size_t tok0 = (size_t)b * L_ + (size_t)c * CH_;

    sExp[tid] = expf(cumA_ws[(size_t)blockIdx.x * CH_ + tid]);
    __syncthreads();

    const unsigned short* Sp = Sscan + (size_t)blockIdx.x * (HD * DS_);   // (p,n) bf16
    for (int ltile = wv; ltile < 16; ltile += 8) {
        const unsigned short* Crow = cg + (tok0 + ltile * 16 + lm) * DS_ + hl * 8;
        Frag a0, a1;
        a0.q[0] = *(const u32x4*)(Crow);
        a0.q[1] = *(const u32x4*)(Crow + 16);
        a1.q[0] = *(const u32x4*)(Crow + 32);
        a1.q[1] = *(const u32x4*)(Crow + 48);
        for (int pg = 0; pg < 2; ++pg) {
            v8f acc[8] = {};
#pragma unroll
            for (int k0 = 0; k0 < DS_; k0 += 32) {
                const Frag& a = (k0 == 0) ? a0 : a1;
#pragma unroll
                for (int j = 0; j < 8; ++j) {
                    const int p = (pg * 8 + j) * 16 + lm;
                    const unsigned short* Srow = Sp + (size_t)p * DS_ + k0 + hl * 8;
                    Frag bb;
                    bb.q[0] = *(const u32x4*)(Srow);
                    bb.q[1] = *(const u32x4*)(Srow + 16);
                    acc[j] = wmma_bf16(a, bb, acc[j]);
                }
            }
#pragma unroll
            for (int j = 0; j < 8; ++j) {
                const int p = (pg * 8 + j) * 16 + lm;
#pragma unroll
                for (int r = 0; r < 8; ++r) {
                    const int ll = ltile * 16 + r + hl * 8;
                    const size_t yi = (tok0 + ll) * DI + (size_t)h * HD + p;
                    float y = bf16_to_f32(ybf[yi]) + acc[j][r] * sExp[ll];
                    float z = bf16_to_f32(zbf[yi]);
                    ybf[yi] = f32_to_bf16(y * (z / (1.f + expf(-z))));
                }
            }
        }
    }
}

// ---------------- RMS norm per token ----------------
__global__ __launch_bounds__(256) void rmsnorm_kernel(const unsigned short* __restrict__ ybf,
                                                      const float* __restrict__ gammav,
                                                      unsigned short* __restrict__ ynorm) {
    __shared__ float red[8];
    __shared__ float sScale;
    const int t = threadIdx.x;
    const size_t tok = blockIdx.x;
    const unsigned short* row = ybf + tok * DI;
    float ss = 0.f;
    for (int i = t; i < DI; i += 256) { float v = bf16_to_f32(row[i]); ss += v * v; }
    for (int off = 16; off; off >>= 1) ss += __shfl_xor(ss, off, 32);
    if ((t & 31) == 0) red[t >> 5] = ss;
    __syncthreads();
    if (t == 0) {
        float s2 = 0.f;
        for (int i = 0; i < 8; ++i) s2 += red[i];
        sScale = rsqrtf(s2 / (float)DI + 1e-5f);
    }
    __syncthreads();
    const float sc = sScale;
    unsigned short* orow = ynorm + tok * DI;
    for (int i = t; i < DI; i += 256)
        orow[i] = f32_to_bf16(bf16_to_f32(row[i]) * sc * gammav[i]);
}

// ---------------- host launcher ----------------
extern "C" void kernel_launch(void* const* d_in, const int* in_sizes, int n_in,
                              void* d_out, int out_size, void* d_ws, size_t ws_size,
                              hipStream_t stream)
{
    (void)in_sizes; (void)n_in; (void)out_size; (void)ws_size;
    const float* inputs = (const float*)d_in[0];
    const float* Wz     = (const float*)d_in[1];
    const float* Wxbc   = (const float*)d_in[2];
    const float* Wdt    = (const float*)d_in[3];
    const float* convk  = (const float*)d_in[4];
    const float* dtb    = (const float*)d_in[5];
    const float* alog   = (const float*)d_in[6];
    const float* gma    = (const float*)d_in[7];
    const float* Wout   = (const float*)d_in[8];
    float* out = (float*)d_out;

    char* ws = (char*)d_ws;
    size_t off = 0;
    auto take = [&](size_t bytes) { char* p = ws + off; off += (bytes + 255) & ~(size_t)255; return p; };

    unsigned short* in_bf  = (unsigned short*)take((size_t)TOKS * DM * 2);
    unsigned short* Wz_t   = (unsigned short*)take((size_t)DI * DM * 2);
    unsigned short* Wxbc_t = (unsigned short*)take((size_t)CONVD * DM * 2);
    unsigned short* Wout_t = (unsigned short*)take((size_t)DM * DI * 2);
    float*          dtw    = (float*)take((size_t)TOKS * NH * 4);
    unsigned short* z_bf   = (unsigned short*)take((size_t)TOKS * DI * 2);
    unsigned short* xbc_bf = (unsigned short*)take((size_t)TOKS * CONVD * 2);  // reused as ynorm
    unsigned short* xdt_bf = (unsigned short*)take((size_t)B_ * NH * HD * L_ * 2);
    unsigned short* bg     = (unsigned short*)take((size_t)TOKS * DS_ * 2);
    unsigned short* bt     = (unsigned short*)take((size_t)TOKS * DS_ * 2);
    unsigned short* cgp    = (unsigned short*)take((size_t)TOKS * DS_ * 2);
    unsigned short* y_bf   = (unsigned short*)take((size_t)TOKS * DI * 2);
    float*          Sloc   = (float*)take((size_t)512 * HD * DS_ * 4);
    unsigned short* Sscan  = (unsigned short*)take((size_t)512 * HD * DS_ * 2);
    float*          cumA   = (float*)take((size_t)512 * CH_ * 4);
    float*          Asum   = (float*)take((size_t)512 * 4);
    unsigned short* ynorm  = xbc_bf;   // xbc dead after conv kernel; safe reuse

    // conversions / transposes
    cvt_bf16_kernel<<<(TOKS * DM + 255) / 256, 256, 0, stream>>>(inputs, in_bf, (size_t)TOKS * DM);
    transpose_bf16_kernel<<<(DM * DI + 255) / 256, 256, 0, stream>>>(Wz, Wz_t, DM, DI);
    transpose_bf16_kernel<<<(DM * CONVD + 255) / 256, 256, 0, stream>>>(Wxbc, Wxbc_t, DM, CONVD);
    transpose_bf16_kernel<<<(DI * DM + 255) / 256, 256, 0, stream>>>(Wout, Wout_t, DI, DM);

    dt_kernel<<<TOKS, 256, 0, stream>>>(inputs, Wdt, dtb, dtw);

    // input projections (WMMA)
    gemm_bf16<false><<<(TOKS / 16) * (DI / 64) / 8, 256, 0, stream>>>(in_bf, Wz_t, z_bf, TOKS, DI, DM);
    gemm_bf16<false><<<(TOKS / 16) * (CONVD / 64) / 8, 256, 0, stream>>>(in_bf, Wxbc_t, xbc_bf, TOKS, CONVD, DM);

    // conv + SiLU + routing (also applies dt scaling and transposes x)
    conv_silu_route_kernel<<<(int)(((size_t)TOKS * CONVD + 255) / 256), 256, 0, stream>>>(
        xbc_bf, convk, dtw, xdt_bf, bg, bt, cgp);

    // chunked SSD (WMMA): y_diag + local states
    ssd_chunk_kernel<<<512, 256, 0, stream>>>(dtw, alog, bt, bg, cgp, xdt_bf, y_bf, Sloc, cumA, Asum);

    // sequential inter-chunk recurrence
    state_scan_kernel<<<(B_ * NH * HD * DS_ + 255) / 256, 256, 0, stream>>>(Sloc, Sscan, Asum);

    // y_off (WMMA) + gating
    yoff_gate_kernel<<<512, 256, 0, stream>>>(cgp, Sscan, cumA, z_bf, y_bf);

    // RMS norm
    rmsnorm_kernel<<<TOKS, 256, 0, stream>>>(y_bf, gma, ynorm);

    // output projection (WMMA, f32 out)
    gemm_bf16<true><<<(TOKS / 16) * (DM / 64) / 8, 256, 0, stream>>>(ynorm, Wout_t, out, TOKS, DM, DI);
}
